// GATNet_15942918603215
// MI455X (gfx1250) — compile-verified, matrix operand order checked by reading, not run on hardware
//
#include <hip/hip_runtime.h>
#include <cmath>

// ---------------- problem constants (match reference) ----------------
#define NNODES 50000
#define NEDGES 500000
#define ETOT   (NEDGES + NNODES)   // edges + self loops
#define INCH   256
#define HIDC   64
#define NHEADS 8
#define HD     512                 // NHEADS * HIDC
#define OUTC   16
#define NEG_SLOPE 0.2f

typedef __attribute__((ext_vector_type(16))) __bf16 v16bf;
typedef __attribute__((ext_vector_type(8)))  float  v8f;

// ---------------- helpers ----------------
__device__ __forceinline__ unsigned enc_f32(float f) {
  unsigned u = __float_as_uint(f);
  return (u & 0x80000000u) ? ~u : (u | 0x80000000u);
}
__device__ __forceinline__ float dec_f32(unsigned u) {
  u = (u & 0x80000000u) ? (u & 0x7fffffffu) : ~u;
  return __uint_as_float(u);
}
// split f32 -> bf16 hi + bf16 lo (error-compensated WMMA inputs)
__device__ __forceinline__ void split_bf16(const float* f, v16bf& hi, v16bf& lo) {
#pragma unroll
  for (int j = 0; j < 16; j++) {
    float v = f[j];
    __bf16 h = (__bf16)v;
    hi[j] = h;
    lo[j] = (__bf16)(v - (float)h);
  }
}

// ---------------- init kernels ----------------
__global__ void fill_f32(float* p, float v, int n) {
  int i = blockIdx.x * blockDim.x + threadIdx.x;
  if (i < n) p[i] = v;
}
__global__ void fill_u32(unsigned* p, unsigned v, int n) {
  int i = blockIdx.x * blockDim.x + threadIdx.x;
  if (i < n) p[i] = v;
}
__global__ void fill_bias(float* p, const float* __restrict__ b, int n, int C) {
  int i = blockIdx.x * blockDim.x + threadIdx.x;
  if (i < n) p[i] = b[i % C];
}

// ---------------- GEMM1: h1 = x @ W1 + b1  (WMMA bf16 hi/lo split) ----------------
// wave -> (mtile, ngroup of 64 cols); K = 256 in 8 steps of 32.
__global__ __launch_bounds__(256) void gemm1_wmma(
    const float* __restrict__ x, const float* __restrict__ W,
    const float* __restrict__ bias, float* __restrict__ h) {
  const int wave = (blockIdx.x * blockDim.x + threadIdx.x) >> 5;
  const int lane = threadIdx.x & 31;
  const int mtile = wave >> 3;
  const int ng    = wave & 7;
  if (mtile >= NNODES / 16) return;
  const int half = lane >> 4;
  const int m    = lane & 15;
  const int row  = mtile * 16 + m;

  v8f zero = {0.f, 0.f, 0.f, 0.f, 0.f, 0.f, 0.f, 0.f};
  v8f acc[4] = {zero, zero, zero, zero};

  const float* xr = x + (size_t)row * INCH;
  for (int k0 = 0; k0 < INCH; k0 += 32) {
    const int base = k0 + 8 * half;
    float fa[16];
#pragma unroll
    for (int j = 0; j < 8; j++) fa[j]     = xr[base + j];
#pragma unroll
    for (int j = 0; j < 8; j++) fa[8 + j] = xr[base + 16 + j];
    v16bf ahi, alo; split_bf16(fa, ahi, alo);
#pragma unroll
    for (int t = 0; t < 4; t++) {
      const int ncol = ng * 64 + t * 16 + m;
      float fb[16];
#pragma unroll
      for (int j = 0; j < 8; j++) fb[j]     = W[(size_t)(base + j) * HD + ncol];
#pragma unroll
      for (int j = 0; j < 8; j++) fb[8 + j] = W[(size_t)(base + 16 + j) * HD + ncol];
      v16bf bhi, blo; split_bf16(fb, bhi, blo);
      acc[t] = __builtin_amdgcn_wmma_f32_16x16x32_bf16(false, ahi, false, bhi, (short)0, acc[t], false, false);
      acc[t] = __builtin_amdgcn_wmma_f32_16x16x32_bf16(false, ahi, false, blo, (short)0, acc[t], false, false);
      acc[t] = __builtin_amdgcn_wmma_f32_16x16x32_bf16(false, alo, false, bhi, (short)0, acc[t], false, false);
    }
  }
#pragma unroll
  for (int t = 0; t < 4; t++) {
    const int ncol = ng * 64 + t * 16 + m;
    const float bv = bias[ncol];
#pragma unroll
    for (int r = 0; r < 8; r++) {
      const int orow = mtile * 16 + r + 8 * half;
      h[(size_t)orow * HD + ncol] = acc[t][r] + bv;
    }
  }
}

// ---------------- GEMM2: h2 = elu(out1) @ W2 + b2 (out1 pre-ELU'd in place) ----------------
__global__ __launch_bounds__(256) void gemm2_wmma(
    const float* __restrict__ a, const float* __restrict__ W,
    const float* __restrict__ bias, float* __restrict__ h) {
  const int wave = (blockIdx.x * blockDim.x + threadIdx.x) >> 5;
  const int lane = threadIdx.x & 31;
  if (wave >= NNODES / 16) return;
  const int half = lane >> 4;
  const int m    = lane & 15;
  const int row  = wave * 16 + m;

  v8f acc = {0.f, 0.f, 0.f, 0.f, 0.f, 0.f, 0.f, 0.f};
  const float* ar = a + (size_t)row * HD;
  for (int k0 = 0; k0 < HD; k0 += 32) {
    const int base = k0 + 8 * half;
    float fa[16];
#pragma unroll
    for (int j = 0; j < 8; j++) fa[j]     = ar[base + j];
#pragma unroll
    for (int j = 0; j < 8; j++) fa[8 + j] = ar[base + 16 + j];
    v16bf ahi, alo; split_bf16(fa, ahi, alo);
    float fb[16];
#pragma unroll
    for (int j = 0; j < 8; j++) fb[j]     = W[(size_t)(base + j) * OUTC + m];
#pragma unroll
    for (int j = 0; j < 8; j++) fb[8 + j] = W[(size_t)(base + 16 + j) * OUTC + m];
    v16bf bhi, blo; split_bf16(fb, bhi, blo);
    acc = __builtin_amdgcn_wmma_f32_16x16x32_bf16(false, ahi, false, bhi, (short)0, acc, false, false);
    acc = __builtin_amdgcn_wmma_f32_16x16x32_bf16(false, ahi, false, blo, (short)0, acc, false, false);
    acc = __builtin_amdgcn_wmma_f32_16x16x32_bf16(false, alo, false, bhi, (short)0, acc, false, false);
  }
  const float bv = bias[m];
#pragma unroll
  for (int r = 0; r < 8; r++) {
    const int orow = wave * 16 + r + 8 * half;
    h[(size_t)orow * OUTC + m] = acc[r] + bv;
  }
}

// ---------------- elementwise ELU in place ----------------
__global__ void elu_inplace(float* p, int n) {
  int i = blockIdx.x * blockDim.x + threadIdx.x;
  if (i < n) {
    float v = p[i];
    p[i] = v > 0.f ? v : expm1f(v);
  }
}

// ---------------- per-node attention logits, layer 1 ----------------
__global__ void att_logits1(const float* __restrict__ h,
                            const float* __restrict__ As, const float* __restrict__ Ad,
                            float* __restrict__ as, float* __restrict__ ad) {
  int i = blockIdx.x * blockDim.x + threadIdx.x;
  if (i >= NNODES * NHEADS) return;
  int node = i / NHEADS, hd = i % NHEADS;
  const float* hp = h + (size_t)node * HD + hd * HIDC;
  const float* sp = As + hd * HIDC;
  const float* dp = Ad + hd * HIDC;
  float s = 0.f, d = 0.f;
#pragma unroll 4
  for (int c = 0; c < HIDC; c++) { s += hp[c] * sp[c]; d += hp[c] * dp[c]; }
  as[i] = s; ad[i] = d;
}

// ---------------- per-node attention logits, layer 2 (1 head, 16ch) ----------------
__global__ void att_logits2(const float* __restrict__ h,
                            const float* __restrict__ As, const float* __restrict__ Ad,
                            float* __restrict__ as, float* __restrict__ ad) {
  int n = blockIdx.x * blockDim.x + threadIdx.x;
  if (n >= NNODES) return;
  const float* hp = h + (size_t)n * OUTC;
  float s = 0.f, d = 0.f;
#pragma unroll
  for (int c = 0; c < OUTC; c++) { s += hp[c] * As[c]; d += hp[c] * Ad[c]; }
  as[n] = s; ad[n] = d;
}

__device__ __forceinline__ void edge_ends(const int* __restrict__ ei, int e, int& src, int& dst) {
  if (e < NEDGES) { src = ei[e]; dst = ei[NEDGES + e]; }
  else            { src = dst = e - NEDGES; }
}

// ---------------- layer-1 edge passes (8 heads) ----------------
__global__ void edge_score_max1(const int* __restrict__ ei,
                                const float* __restrict__ as, const float* __restrict__ ad,
                                float* __restrict__ wbuf, unsigned* __restrict__ emax) {
  int i = blockIdx.x * blockDim.x + threadIdx.x;
  if (i >= ETOT * NHEADS) return;
  int e = i / NHEADS, hd = i % NHEADS;
  int src, dst; edge_ends(ei, e, src, dst);
  float sc = as[(size_t)src * NHEADS + hd] + ad[(size_t)dst * NHEADS + hd];
  sc = sc > 0.f ? sc : NEG_SLOPE * sc;
  wbuf[i] = sc;
  atomicMax(&emax[(size_t)dst * NHEADS + hd], enc_f32(sc));
}
__global__ void edge_expsum1(const int* __restrict__ ei,
                             const unsigned* __restrict__ emax,
                             float* __restrict__ wbuf, float* __restrict__ denom) {
  int i = blockIdx.x * blockDim.x + threadIdx.x;
  if (i >= ETOT * NHEADS) return;
  int e = i / NHEADS, hd = i % NHEADS;
  int src, dst; edge_ends(ei, e, src, dst);
  (void)src;
  float w = expf(wbuf[i] - dec_f32(emax[(size_t)dst * NHEADS + hd]));
  wbuf[i] = w;
  atomicAdd(&denom[(size_t)dst * NHEADS + hd], w);
}
// one wave per edge: lane l covers channels l, l+32, ... (coalesced gather + f32 atomics)
__global__ __launch_bounds__(256) void edge_agg1(const int* __restrict__ ei,
                                                 const float* __restrict__ h1,
                                                 const float* __restrict__ wbuf,
                                                 const float* __restrict__ denom,
                                                 float* __restrict__ out) {
  int e = (blockIdx.x * blockDim.x + threadIdx.x) >> 5;
  if (e >= ETOT) return;
  int lane = threadIdx.x & 31;
  int src, dst; edge_ends(ei, e, src, dst);
  float alpha[NHEADS];
#pragma unroll
  for (int hd = 0; hd < NHEADS; hd++)
    alpha[hd] = wbuf[(size_t)e * NHEADS + hd] / (denom[(size_t)dst * NHEADS + hd] + 1e-16f);
  const float* hs = h1 + (size_t)src * HD;
  float* od = out + (size_t)dst * HD;
#pragma unroll
  for (int j = 0; j < 16; j++) {
    int ch = lane + 32 * j;
    atomicAdd(&od[ch], alpha[ch >> 6] * hs[ch]);
  }
}

// ---------------- layer-2 edge passes (1 head, 16 channels) ----------------
__global__ void edge_score_max2(const int* __restrict__ ei,
                                const float* __restrict__ as, const float* __restrict__ ad,
                                float* __restrict__ wbuf, unsigned* __restrict__ emax) {
  int e = blockIdx.x * blockDim.x + threadIdx.x;
  if (e >= ETOT) return;
  int src, dst; edge_ends(ei, e, src, dst);
  float sc = as[src] + ad[dst];
  sc = sc > 0.f ? sc : NEG_SLOPE * sc;
  wbuf[e] = sc;
  atomicMax(&emax[dst], enc_f32(sc));
}
__global__ void edge_expsum2(const int* __restrict__ ei,
                             const unsigned* __restrict__ emax,
                             float* __restrict__ wbuf, float* __restrict__ denom) {
  int e = blockIdx.x * blockDim.x + threadIdx.x;
  if (e >= ETOT) return;
  int src, dst; edge_ends(ei, e, src, dst);
  (void)src;
  float w = expf(wbuf[e] - dec_f32(emax[dst]));
  wbuf[e] = w;
  atomicAdd(&denom[dst], w);
}
__global__ void edge_agg2(const int* __restrict__ ei,
                          const float* __restrict__ h2,
                          const float* __restrict__ wbuf,
                          const float* __restrict__ denom,
                          float* __restrict__ out) {
  int i = blockIdx.x * blockDim.x + threadIdx.x;
  if (i >= ETOT * OUTC) return;
  int e = i / OUTC, c = i % OUTC;
  int src, dst; edge_ends(ei, e, src, dst);
  float alpha = wbuf[e] / (denom[dst] + 1e-16f);
  atomicAdd(&out[(size_t)dst * OUTC + c], alpha * h2[(size_t)src * OUTC + c]);
}

// ---------------- final log_softmax over 16 classes ----------------
__global__ void log_softmax16(const float* __restrict__ in, float* __restrict__ out) {
  int n = blockIdx.x * blockDim.x + threadIdx.x;
  if (n >= NNODES) return;
  const float* p = in + (size_t)n * OUTC;
  float v[OUTC], mx = -3.402823466e38f;
#pragma unroll
  for (int c = 0; c < OUTC; c++) { v[c] = p[c]; mx = v[c] > mx ? v[c] : mx; }
  float s = 0.f;
#pragma unroll
  for (int c = 0; c < OUTC; c++) s += expf(v[c] - mx);
  float lse = mx + logf(s);
  float* o = out + (size_t)n * OUTC;
#pragma unroll
  for (int c = 0; c < OUTC; c++) o[c] = v[c] - lse;
}

// ---------------- workspace layout (bytes) ----------------
constexpr size_t OFF_H1    = 0;
constexpr size_t OFF_OUT1  = OFF_H1    + (size_t)NNODES * HD * 4;      // 102.4 MB
constexpr size_t OFF_H2    = OFF_OUT1  + (size_t)NNODES * HD * 4;      // 102.4 MB
constexpr size_t OFF_AS1   = OFF_H2    + (size_t)NNODES * OUTC * 4;
constexpr size_t OFF_AD1   = OFF_AS1   + (size_t)NNODES * NHEADS * 4;
constexpr size_t OFF_EMAX1 = OFF_AD1   + (size_t)NNODES * NHEADS * 4;
constexpr size_t OFF_DEN1  = OFF_EMAX1 + (size_t)NNODES * NHEADS * 4;
constexpr size_t OFF_W1B   = OFF_DEN1  + (size_t)NNODES * NHEADS * 4;
constexpr size_t OFF_OUT2  = OFF_W1B   + (size_t)ETOT * NHEADS * 4;    // 17.6 MB
constexpr size_t OFF_AS2   = OFF_OUT2  + (size_t)NNODES * OUTC * 4;
constexpr size_t OFF_AD2   = OFF_AS2   + (size_t)NNODES * 4;
constexpr size_t OFF_EMAX2 = OFF_AD2   + (size_t)NNODES * 4;
constexpr size_t OFF_DEN2  = OFF_EMAX2 + (size_t)NNODES * 4;
constexpr size_t OFF_W2B   = OFF_DEN2  + (size_t)NNODES * 4;

static inline int nblk(long long n, int t) { return (int)((n + t - 1) / t); }

extern "C" void kernel_launch(void* const* d_in, const int* in_sizes, int n_in,
                              void* d_out, int out_size, void* d_ws, size_t ws_size,
                              hipStream_t stream) {
  const float* x     = (const float*)d_in[0];
  const int*   ei    = (const int*)d_in[1];
  const float* W1    = (const float*)d_in[2];
  const float* Asrc1 = (const float*)d_in[3];
  const float* Adst1 = (const float*)d_in[4];
  const float* b1    = (const float*)d_in[5];
  const float* W2    = (const float*)d_in[6];
  const float* Asrc2 = (const float*)d_in[7];
  const float* Adst2 = (const float*)d_in[8];
  const float* b2    = (const float*)d_in[9];
  float* out = (float*)d_out;

  char* ws = (char*)d_ws;
  float*    h1    = (float*)(ws + OFF_H1);
  float*    out1  = (float*)(ws + OFF_OUT1);
  float*    h2    = (float*)(ws + OFF_H2);
  float*    as1   = (float*)(ws + OFF_AS1);
  float*    ad1   = (float*)(ws + OFF_AD1);
  unsigned* emax1 = (unsigned*)(ws + OFF_EMAX1);
  float*    den1  = (float*)(ws + OFF_DEN1);
  float*    w1b   = (float*)(ws + OFF_W1B);
  float*    out2  = (float*)(ws + OFF_OUT2);
  float*    as2   = (float*)(ws + OFF_AS2);
  float*    ad2   = (float*)(ws + OFF_AD2);
  unsigned* emax2 = (unsigned*)(ws + OFF_EMAX2);
  float*    den2  = (float*)(ws + OFF_DEN2);
  float*    w2b   = (float*)(ws + OFF_W2B);

  const int T = 256;

  // ---- init accumulators ----
  fill_bias<<<nblk((long long)NNODES * HD, T), T, 0, stream>>>(out1, b1, NNODES * HD, HD);
  fill_u32 <<<nblk(NNODES * NHEADS, T), T, 0, stream>>>(emax1, 0u, NNODES * NHEADS);
  fill_f32 <<<nblk(NNODES * NHEADS, T), T, 0, stream>>>(den1, 0.f, NNODES * NHEADS);
  fill_bias<<<nblk((long long)NNODES * OUTC, T), T, 0, stream>>>(out2, b2, NNODES * OUTC, OUTC);
  fill_u32 <<<nblk(NNODES, T), T, 0, stream>>>(emax2, 0u, NNODES);
  fill_f32 <<<nblk(NNODES, T), T, 0, stream>>>(den2, 0.f, NNODES);

  // ---- layer 1 ----
  // GEMM1: 3125 m-tiles * 8 n-groups = 25000 waves, 8 waves/block
  gemm1_wmma<<<(NNODES / 16) * 8 / 8, T, 0, stream>>>(x, W1, b1, h1);
  att_logits1<<<nblk(NNODES * NHEADS, T), T, 0, stream>>>(h1, Asrc1, Adst1, as1, ad1);
  edge_score_max1<<<nblk((long long)ETOT * NHEADS, T), T, 0, stream>>>(ei, as1, ad1, w1b, emax1);
  edge_expsum1  <<<nblk((long long)ETOT * NHEADS, T), T, 0, stream>>>(ei, emax1, w1b, den1);
  edge_agg1     <<<nblk((long long)ETOT * 32, T), T, 0, stream>>>(ei, h1, w1b, den1, out1);
  elu_inplace   <<<nblk((long long)NNODES * HD, T), T, 0, stream>>>(out1, NNODES * HD);

  // ---- layer 2 ----
  gemm2_wmma<<<nblk(NNODES / 16, 8) , T, 0, stream>>>(out1, W2, b2, h2);
  att_logits2<<<nblk(NNODES, T), T, 0, stream>>>(h2, Asrc2, Adst2, as2, ad2);
  edge_score_max2<<<nblk(ETOT, T), T, 0, stream>>>(ei, as2, ad2, w2b, emax2);
  edge_expsum2  <<<nblk(ETOT, T), T, 0, stream>>>(ei, emax2, w2b, den2);
  edge_agg2     <<<nblk((long long)ETOT * OUTC, T), T, 0, stream>>>(ei, h2, w2b, den2, out2);

  // ---- output ----
  log_softmax16<<<nblk(NNODES, T), T, 0, stream>>>(out2, out);

  (void)in_sizes; (void)n_in; (void)out_size; (void)ws_size;
}
